// Deform_resnet_48232482734881
// MI455X (gfx1250) — compile-verified, matrix operand order checked by reading, not run on hardware
//
#include <hip/hip_runtime.h>

typedef _Float16 h16_t;
typedef __attribute__((ext_vector_type(16))) _Float16 v16h;
typedef __attribute__((ext_vector_type(8)))  float    v8f;

#define BN_EPS 1e-5f
#define GEMM_WAVES 8

union H2    { h16_t h[2]; unsigned u; };
union AFrag { unsigned u[8]; v16h v; };
union BFrag { uint4 q[2]; unsigned u[8]; v16h v; };
union CFrag { float f[8]; v8f v; };

// ---------------------------------------------------------------------------
// B-fragment scatter index.
// B is stored fragment-ready: for k-tile kt (32 K values) and n-tile nt
// (16 N values), the 32 lanes each own 8 contiguous uints (16 f16).
// Mapping (matches ISA 16-bit A/B VGPR layout):
//   kr = k % 32 (pair-aligned), g = kr>>3, hi = g&1,
//   v2 = (g>>1)*4 + ((kr>>1)&3), lane = (n%16) + 16*hi.
// ---------------------------------------------------------------------------
__device__ __forceinline__ long bfrag_idx(int q, long n, long ntiles) {
  int kt  = q >> 4;              // k-tile (q = pair index, 16 pairs per tile)
  int kr  = (q << 1) & 31;       // k within tile (even)
  int g   = kr >> 3;
  int hi  = g & 1;
  int v2  = ((g >> 1) << 2) | ((kr >> 1) & 3);
  int ln  = (int)(n & 15) + (hi << 4);
  return ((((long)kt * ntiles + (n >> 4)) * 32 + ln) << 3) + v2;
}

// ---------------------------------------------------------------------------
// Weight conversion: fp32 -> packed f16 pairs, K padded to Kp (=2*Kp2).
// A stays in plain row-major pair layout (it goes through LDS).
// ---------------------------------------------------------------------------
__global__ void dcn_cvt_pm_k(const float* __restrict__ wp,
                             const float* __restrict__ wm,
                             unsigned* __restrict__ A, int K, int Kp2) {
  long tid = (long)blockIdx.x * 256 + threadIdx.x;
  long total = 32L * Kp2;
  if (tid >= total) return;
  int m = (int)(tid / Kp2);
  int q = (int)(tid % Kp2);
  H2 o;
  #pragma unroll
  for (int e = 0; e < 2; ++e) {
    int k = 2 * q + e;
    float v = 0.f;
    if (k < K) {
      if (m < 18)      v = wp[(long)m * K + k];
      else if (m < 27) v = wm[(long)(m - 18) * K + k];
    }
    o.h[e] = (h16_t)v;
  }
  A[tid] = o.u;
}

__global__ void dcn_cvt_w_k(const float* __restrict__ wc,
                            unsigned* __restrict__ A, int M, int K, int Kp2) {
  long tid = (long)blockIdx.x * 256 + threadIdx.x;
  long total = (long)M * Kp2;
  if (tid >= total) return;
  int m = (int)(tid / Kp2);
  int q = (int)(tid % Kp2);
  H2 o;
  #pragma unroll
  for (int e = 0; e < 2; ++e) {
    int k = 2 * q + e;
    o.h[e] = (h16_t)((k < K) ? wc[(long)m * K + k] : 0.f);
  }
  A[tid] = o.u;
}

// ---------------------------------------------------------------------------
// im2col for the offset/mask conv (plain 3x3, pad 1, stride s) -> B fragments.
// Activations use layout [C][B][H][W].
// ---------------------------------------------------------------------------
__global__ void dcn_im2col_k(const float* __restrict__ x, unsigned* __restrict__ col,
                             int C, int B, int H, int W, int oh, int ow, int s,
                             int K, int Kp2, long Npos) {
  long tid = (long)blockIdx.x * 256 + threadIdx.x;
  if (tid >= (long)Kp2 * Npos) return;
  long n = tid % Npos;
  int  q = (int)(tid / Npos);
  int bi = (int)(n / (oh * ow));
  int rem = (int)(n % (oh * ow));
  int i = rem / ow, j = rem % ow;
  H2 o;
  #pragma unroll
  for (int e = 0; e < 2; ++e) {
    int k = 2 * q + e;
    float v = 0.f;
    if (k < K) {
      int ic = k / 9, t = k % 9;
      int y  = i * s + t / 3 - 1;
      int xx = j * s + t % 3 - 1;
      if (y >= 0 && y < H && xx >= 0 && xx < W)
        v = x[(((long)ic * B + bi) * H + y) * W + xx];
    }
    o.h[e] = (h16_t)v;
  }
  col[bfrag_idx(q, n, Npos >> 4)] = o.u;
}

// ---------------------------------------------------------------------------
// Bilinear sampling + modulation -> main-conv B fragments (DCNv2 semantics).
// om layout: rows 0..8 = dy, 9..17 = dx, 18..26 = sigmoid mask, each [Npos].
// ---------------------------------------------------------------------------
__global__ void dcn_sample_k(const float* __restrict__ x, const float* __restrict__ om,
                             unsigned* __restrict__ col,
                             int C, int B, int H, int W, int oh, int ow, int s,
                             int K, int Kp2, long Npos) {
  long tid = (long)blockIdx.x * 256 + threadIdx.x;
  if (tid >= (long)Kp2 * Npos) return;
  long n = tid % Npos;
  int  q = (int)(tid / Npos);
  int bi = (int)(n / (oh * ow));
  int rem = (int)(n % (oh * ow));
  int i = rem / ow, j = rem % ow;
  int Hp = H + 2, Wp = W + 2;
  H2 o;
  #pragma unroll
  for (int e = 0; e < 2; ++e) {
    int k = 2 * q + e;
    float val = 0.f;
    if (k < K) {
      int ic = k / 9, t = k % 9;
      float offy = om[(long)t * Npos + n];
      float offx = om[(long)(9 + t) * Npos + n];
      float mv   = om[(long)(18 + t) * Npos + n];
      float py = offy + (float)(t / 3 - 1) + (float)(1 + i * s);
      float px = offx + (float)(t % 3 - 1) + (float)(1 + j * s);
      float fy = floorf(py), fx = floorf(px);
      float hmax = (float)(Hp - 1), wmax = (float)(Wp - 1);
      float qy0 = fminf(fmaxf(fy, 0.f), hmax);
      float qx0 = fminf(fmaxf(fx, 0.f), wmax);
      float qy1 = fminf(fmaxf(fy + 1.f, 0.f), hmax);
      float qx1 = fminf(fmaxf(fx + 1.f, 0.f), wmax);
      float pyc = fminf(fmaxf(py, 0.f), hmax);
      float pxc = fminf(fmaxf(px, 0.f), wmax);
      float glt = (1.f + (qy0 - pyc)) * (1.f + (qx0 - pxc));
      float grb = (1.f - (qy1 - pyc)) * (1.f - (qx1 - pxc));
      float glb = (1.f + (qy0 - pyc)) * (1.f - (qx1 - pxc));
      float grt = (1.f - (qy1 - pyc)) * (1.f + (qx0 - pxc));
      const float* xc = x + ((long)ic * B + bi) * H * W;
      auto samp = [&](float qy, float qx) -> float {
        int iy = (int)qy, ix = (int)qx;  // padded coords; pad ring is zero
        if (iy >= 1 && iy <= H && ix >= 1 && ix <= W)
          return xc[(long)(iy - 1) * W + (ix - 1)];
        return 0.f;
      };
      val = glt * samp(qy0, qx0) + grb * samp(qy1, qx1) +
            glb * samp(qy0, qx1) + grt * samp(qy1, qx0);
      val *= mv;
    }
    o.h[e] = (h16_t)val;
  }
  col[bfrag_idx(q, n, Npos >> 4)] = o.u;
}

// ---------------------------------------------------------------------------
// WMMA GEMM: C[M][N] = A[M][Kp] @ B[Kp][N] (f16 in, f32 acc).
// A: row-major packed pairs (stride Kp/2 uints), double-buffered through LDS.
// B: fragment-ready (see bfrag_idx).
// 8 waves/block; each wave computes a 16x32 C tile (2 n-tiles, shared A frag,
// 2 WMMAs per K-step).  Per-lane B pointer advances by a constant stride;
// second tile reached via immediate offsets (bp[64],bp[65]).
// Grids divide exactly (all N are multiples of 256) -> no guards, EXEC all-1s.
// mode 0: offset/mask epilogue (bias, sigmoid for mask rows, rows>=27 dropped)
// mode 1: BN (+residual) (+ReLU) epilogue.
// ---------------------------------------------------------------------------
__global__ __launch_bounds__(256) void dcn_wmma_gemm_k(
    const unsigned* __restrict__ A, const unsigned* __restrict__ Bf,
    long N, int Kp, int mode,
    const float* __restrict__ p0, const float* __restrict__ p1,
    const float* __restrict__ p2, const float* __restrict__ p3,
    const float* __restrict__ res, int relu,
    float* __restrict__ out) {
  __shared__ unsigned ldsA[2][256];  // double-buffered 16x32-half A tile
  int tid = threadIdx.x;
  int wv = tid >> 5, lane = tid & 31;
  int hl = lane & 15, hi = lane >> 4;
  int Kp2 = Kp >> 1;
  int nkt = Kp >> 5;                       // number of 32-wide K tiles
  long ntiles = N >> 4;                    // number of 16-wide N tiles
  long m0 = (long)blockIdx.y * 16;
  long nt0 = ((long)blockIdx.x * GEMM_WAVES + wv) * 2;
  int lr = tid >> 4, lc = tid & 15;

  const unsigned* ap = A + (m0 + lr) * (long)Kp2 + lc;
  const uint4* bp = (const uint4*)Bf + ((nt0 * 32 + lane) << 1);
  long bstep = ntiles << 6;                // uint4 units per K tile

  CFrag acc0, acc1;
  #pragma unroll
  for (int r = 0; r < 8; ++r) { acc0.f[r] = 0.f; acc1.f[r] = 0.f; }

  ldsA[0][tid] = ap[0];
  for (int kt = 0; kt < nkt; ++kt) {
    __syncthreads();
    ldsA[(kt + 1) & 1][tid] = ap[(kt + 1) * 16];   // harmless tail over-read
    const unsigned* la = ldsA[kt & 1];
    AFrag a;
    #pragma unroll
    for (int v2 = 0; v2 < 8; ++v2) {
      int qloc = v2 + ((v2 >= 4) ? 4 : 0) + 4 * hi;  // 2 runs of 4 -> b128 ds
      a.u[v2] = la[hl * 16 + qloc];
    }
    BFrag b0, b1;
    b0.q[0] = bp[0];
    b0.q[1] = bp[1];
    b1.q[0] = bp[64];    // next n-tile: +1024B immediate
    b1.q[1] = bp[65];
    __builtin_prefetch(bp + bstep, 0, 1);          // speculative; drop on fault
    bp += bstep;
    acc0.v = __builtin_amdgcn_wmma_f32_16x16x32_f16(
        false, a.v, false, b0.v, (short)0, acc0.v, false, false);
    acc1.v = __builtin_amdgcn_wmma_f32_16x16x32_f16(
        false, a.v, false, b1.v, (short)0, acc1.v, false, false);
  }

  long col0 = nt0 * 16 + hl;
  #pragma unroll
  for (int r = 0; r < 8; ++r) {
    long mrow = m0 + r + 8 * hi;   // C layout: VGPR r -> M=r (lo) / M=8+r (hi)
    float v0 = acc0.f[r], v1 = acc1.f[r];
    if (mode == 0) {
      if (mrow < 18) {
        out[mrow * N + col0]      = v0 + p0[mrow];          // offsets + b_p
        out[mrow * N + col0 + 16] = v1 + p0[mrow];
      } else if (mrow < 27) {
        float bm = p1[mrow - 18];                           // mask + b_m
        out[mrow * N + col0]      = 1.f / (1.f + expf(-(v0 + bm)));
        out[mrow * N + col0 + 16] = 1.f / (1.f + expf(-(v1 + bm)));
      }
    } else {
      float sc = p0[mrow] * rsqrtf(p3[mrow] + BN_EPS);      // gamma/sqrt(var+eps)
      float mu = p2[mrow], be = p1[mrow];
      float a0 = (v0 - mu) * sc + be;                       // BN
      float a1 = (v1 - mu) * sc + be;
      if (res) { a0 += res[mrow * N + col0]; a1 += res[mrow * N + col0 + 16]; }
      if (relu) { a0 = fmaxf(a0, 0.f); a1 = fmaxf(a1, 0.f); }
      out[mrow * N + col0]      = a0;
      out[mrow * N + col0 + 16] = a1;
    }
  }
}

// ---------------------------------------------------------------------------
// AvgPool2d((4,1)) + FC.  y layout [C=512][B=256][H=8][W=1]; feature
// f = c*2 + g (g = h/4 group), out (256,6).
// ---------------------------------------------------------------------------
__global__ void dcn_pool_fc_k(const float* __restrict__ y,
                              const float* __restrict__ fw,
                              const float* __restrict__ fb,
                              float* __restrict__ out) {
  int tid = blockIdx.x * 256 + threadIdx.x;
  if (tid >= 256 * 6) return;
  int bi = tid / 6, o = tid % 6;
  float sum = fb[o];
  for (int c = 0; c < 512; ++c) {
    const float* yc = y + ((long)c * 256 + bi) * 8;
    float pl0 = 0.25f * (yc[0] + yc[1] + yc[2] + yc[3]);
    float pl1 = 0.25f * (yc[4] + yc[5] + yc[6] + yc[7]);
    sum += pl0 * fw[o * 1024 + c * 2] + pl1 * fw[o * 1024 + c * 2 + 1];
  }
  out[bi * 6 + o] = sum;
}

// ---------------------------------------------------------------------------
// Host orchestration
// ---------------------------------------------------------------------------
struct DCW { const float *wp, *bp, *wm, *bm, *wc; };
struct BNW { const float *g, *b, *m, *v; };

static void deform_layer(hipStream_t stream,
                         const float* x, int C, int B, int H, int W, int s,
                         int outc, const DCW& w, const BNW& bn,
                         const float* res, int relu, float* out,
                         unsigned* Apm, unsigned* Ac, unsigned* col, float* off) {
  int oh = (H + 2 - 3) / s + 1;
  int ow = (W + 2 - 3) / s + 1;
  long Npos = (long)B * oh * ow;          // always a multiple of 256 here
  int K = C * 9;
  int Kp = (K + 31) & ~31;
  int Kp2 = Kp >> 1;

  long t1 = 32L * Kp2;
  dcn_cvt_pm_k<<<(unsigned)((t1 + 255) / 256), 256, 0, stream>>>(w.wp, w.wm, Apm, K, Kp2);
  long t2 = (long)outc * Kp2;
  dcn_cvt_w_k<<<(unsigned)((t2 + 255) / 256), 256, 0, stream>>>(w.wc, Ac, outc, K, Kp2);
  long t3 = (long)Kp2 * Npos;
  dcn_im2col_k<<<(unsigned)((t3 + 255) / 256), 256, 0, stream>>>(
      x, col, C, B, H, W, oh, ow, s, K, Kp2, Npos);
  dim3 g1((unsigned)(Npos / (32 * GEMM_WAVES)), 2);
  dcn_wmma_gemm_k<<<g1, 256, 0, stream>>>(Apm, col, Npos, Kp, 0,
                                          w.bp, w.bm, nullptr, nullptr,
                                          nullptr, 0, off);
  dcn_sample_k<<<(unsigned)((t3 + 255) / 256), 256, 0, stream>>>(
      x, off, col, C, B, H, W, oh, ow, s, K, Kp2, Npos);
  dim3 g2((unsigned)(Npos / (32 * GEMM_WAVES)), outc / 16);
  dcn_wmma_gemm_k<<<g2, 256, 0, stream>>>(Ac, col, Npos, Kp, 1,
                                          bn.g, bn.b, bn.m, bn.v, res, relu, out);
}

extern "C" void kernel_launch(void* const* d_in, const int* in_sizes, int n_in,
                              void* d_out, int out_size, void* d_ws, size_t ws_size,
                              hipStream_t stream) {
  (void)in_sizes; (void)n_in; (void)out_size; (void)ws_size;
  int idx = 0;
  auto P = [&]() { return (const float*)d_in[idx++]; };
  auto getDC = [&]() { DCW w; w.wp = P(); w.bp = P(); w.wm = P(); w.bm = P(); w.wc = P(); return w; };
  auto getBN = [&]() { BNW b; b.g = P(); b.b = P(); b.m = P(); b.v = P(); return b; };

  const float* x = P();                        // (256,1,128,9)
  DCW stem = getDC(); BNW stembn = getBN();
  struct Blk { DCW dc1; BNW bn1; DCW dc2; BNW bn2; DCW dcs; BNW bns; } blk[3];
  for (int i = 0; i < 3; ++i) {
    blk[i].dc1 = getDC(); blk[i].bn1 = getBN();
    blk[i].dc2 = getDC(); blk[i].bn2 = getBN();
    blk[i].dcs = getDC(); blk[i].bns = getBN();
  }
  const float* fw = P();
  const float* fb = P();

  // workspace carve
  char* p = (char*)d_ws;
  auto carve = [&](size_t bytes) { void* r = p; p += (bytes + 255) & ~(size_t)255; return r; };
  const size_t ACT = 5242880;                       // max activation floats (stem out)
  float* actA = (float*)carve(ACT * 4);
  float* actY = (float*)carve(ACT * 4);
  float* actS = (float*)carve(ACT * 4);
  float* actB = (float*)carve(ACT * 4);
  unsigned* col = (unsigned*)carve((size_t)14155776 * 4);  // max (Kp/2)*Npos (b1 dc2)
  float* off = (float*)carve((size_t)27 * 81920 * 4);      // max 27*Npos (stem)
  unsigned* Apm = (unsigned*)carve((size_t)32 * 2304 * 4);
  unsigned* Ac  = (unsigned*)carve((size_t)512 * 2304 * 4);
  (void)carve(1024);                                       // tail pad for over-reads

  // stem: relu(bn(dc(x, s=2)))   -> actA  [64][256][64][5]
  deform_layer(stream, x, 1, 256, 128, 9, 2, 64, stem, stembn,
               nullptr, 1, actA, Apm, Ac, col, off);

  int C = 64, H = 64, W = 5;
  float* bin = actA;
  float* bout = actB;
  const int outcs[3] = {128, 256, 512};
  for (int i = 0; i < 3; ++i) {
    int outc = outcs[i];
    int oh = (H + 2 - 3) / 2 + 1;
    int ow = (W + 2 - 3) / 2 + 1;
    // shortcut: s = bn_s(dcs(x, s=2))
    deform_layer(stream, bin, C, 256, H, W, 2, outc, blk[i].dcs, blk[i].bns,
                 nullptr, 0, actS, Apm, Ac, col, off);
    // y = relu(bn1(dc1(x, s=2)))
    deform_layer(stream, bin, C, 256, H, W, 2, outc, blk[i].dc1, blk[i].bn1,
                 nullptr, 1, actY, Apm, Ac, col, off);
    // out = relu(bn2(dc2(y, s=1)) + s)
    deform_layer(stream, actY, outc, 256, oh, ow, 1, outc, blk[i].dc2, blk[i].bn2,
                 actS, 1, bout, Apm, Ac, col, off);
    C = outc; H = oh; W = ow;
    float* t = bin; bin = bout; bout = t;
  }

  // bin now holds b3 output [512][256][8][1]
  dcn_pool_fc_k<<<6, 256, 0, stream>>>(bin, fw, fb, (float*)d_out);
}
